// TransformerBlockKVMem_10024453669677
// MI455X (gfx1250) — matrix operand order repaired and checked by reading, not compiled, not run on hardware
//
#include <hip/hip_runtime.h>
#include <stdint.h>

#define DEVINL __device__ __forceinline__

typedef __attribute__((ext_vector_type(16))) __bf16 bf16x16;
typedef __attribute__((ext_vector_type(8)))  __bf16 bf16x8;
typedef __attribute__((ext_vector_type(8)))  float  f32x8;

static constexpr int Bb = 4, Nn = 2048, Cc = 1024, Hh = 16, Dd = 64;
static constexpr int INNERn = 2730;
static constexpr int INNER_PAD = 2752;   // 86*32: K-aligned for WMMA tiles
static constexpr int VG_N_PAD = 5504;    // 43*128: N-aligned for WMMA tiles
static constexpr int Mrows = Bb * Nn;    // 8192
static constexpr float EPSf = 1e-5f;

// ---------------- helpers ----------------
DEVINL unsigned short f2bf(float f) {               // RNE f32 -> bf16 bits
  unsigned int u = __float_as_uint(f);
  u += 0x7FFFu + ((u >> 16) & 1u);
  return (unsigned short)(u >> 16);
}

// A-fragment (16x32 bf16, M x K): lane = hf*16+ln holds row M=ln,
// elements [0..7] at K = hf*8 + i, [8..15] at K = 16 + hf*8 + i.
// p points at the lo group; hi group is p+16.
DEVINL bf16x16 frag_a(const unsigned short* p) {
  bf16x8 lo = *(const bf16x8*)p;
  bf16x8 hi = *(const bf16x8*)(p + 16);
  bf16x16 r;
#pragma unroll
  for (int i = 0; i < 8; ++i) { r[i] = lo[i]; r[i + 8] = hi[i]; }
  return r;
}

// B-fragment (32x16 bf16, K x N): lane holds col N=ln, elements
// [0..7] at K = hf*8 + i, [8..15] at K = 16 + hf*8 + i (hi group at p+16).
DEVINL bf16x16 frag_b(const unsigned short* p) {
  bf16x8 lo = *(const bf16x8*)p;
  bf16x8 hi = *(const bf16x8*)(p + 16);
  bf16x16 r;
#pragma unroll
  for (int i = 0; i < 8; ++i) { r[i] = lo[i]; r[i + 8] = hi[i]; }
  return r;
}

DEVINL f32x8 wmma_bf16(bf16x16 a, bf16x16 b, f32x8 c) {
  return __builtin_amdgcn_wmma_f32_16x16x32_bf16(false, a, false, b, (short)0, c,
                                                 false, false);
}

// ------------- f32 -> bf16 convert with optional zero padding --------------
// dst is rowsPad x ldd (bf16); src is rows x cols (f32). Pad cells -> 0.
__global__ void convert_pad_kernel(const float* __restrict__ src,
                                   unsigned short* __restrict__ dst, int rows,
                                   int cols, int ldd, int rowsPad) {
  const size_t total = (size_t)rowsPad * ldd;
  size_t i = (size_t)blockIdx.x * blockDim.x + threadIdx.x;
  const size_t stride = (size_t)gridDim.x * blockDim.x;
  for (; i < total; i += stride) {
    size_t r = i / (size_t)ldd;
    int c = (int)(i - r * ldd);
    dst[i] = ((int)r < rows && c < cols) ? f2bf(src[r * (size_t)cols + c])
                                         : (unsigned short)0;
  }
}

// ---------------- generic WMMA GEMM: C = A(bf16,MxK) @ B(bf16,KxN) ----------
// All of M%128, N%128, K%32 must be 0 (buffers pre-padded) -> branch-free
// staging. mode 0: f32 = acc+bias   mode 2: bf16 = silu(acc+bias)
// mode 3: f32 = res + acc + bias.  bias read clamped to col < nbias.
__global__ __launch_bounds__(256) void gemm_bf16_kernel(
    const unsigned short* __restrict__ A, int lda,
    const unsigned short* __restrict__ Bm, int ldb,
    const float* __restrict__ bias, int nbias, void* __restrict__ Cout, int ldc,
    const float* __restrict__ res, int M, int N, int K, int mode) {
  __shared__ unsigned short ldsA[2 * 128 * 40];   // double-buffered A tile
  __shared__ unsigned short ldsBT[2 * 128 * 40];  // double-buffered B^T tile
  const int tid = threadIdx.x;
  const int wv = tid >> 5, lane = tid & 31;
  const int hf = lane >> 4, ln = lane & 15;
  const int wm = wv >> 2, wn = wv & 3;  // 2 x 4 wave grid
  const int bm = blockIdx.y, bn = blockIdx.x;

  // per-thread staging coordinates (2 chunks of 8 bf16 each for A and B)
  int aRow[2], aK8[2], bKr[2], bN8[2];
#pragma unroll
  for (int c = 0; c < 2; ++c) {
    int chunk = tid * 2 + c;  // 0..511
    aRow[c] = chunk >> 2;     // 0..127
    aK8[c] = (chunk & 3) * 8;
    bKr[c] = chunk >> 4;      // 0..31
    bN8[c] = (chunk & 15) * 8;
  }

  f32x8 acc[4][2];
#pragma unroll
  for (int i = 0; i < 4; ++i)
#pragma unroll
    for (int j = 0; j < 2; ++j)
#pragma unroll
      for (int e = 0; e < 8; ++e) acc[i][j][e] = 0.f;

  uint4 va[2], vb[2];
  auto gload = [&](int kt) {
    const int kk = kt << 5;
#pragma unroll
    for (int c = 0; c < 2; ++c) {
      va[c] = *(const uint4*)(A + (size_t)(bm * 128 + aRow[c]) * lda + kk + aK8[c]);
      vb[c] = *(const uint4*)(Bm + (size_t)(kk + bKr[c]) * ldb + bn * 128 + bN8[c]);
    }
  };
  auto lstore = [&](int buf) {
#pragma unroll
    for (int c = 0; c < 2; ++c) {
      *(uint4*)(ldsA + buf * 5120 + aRow[c] * 40 + aK8[c]) = va[c];
      const unsigned short* t = (const unsigned short*)&vb[c];
#pragma unroll
      for (int j = 0; j < 8; ++j)
        ldsBT[buf * 5120 + (bN8[c] + j) * 40 + bKr[c]] = t[j];
    }
  };

  const int ktiles = K >> 5;
  gload(0);
  lstore(0);
  __syncthreads();
  int cur = 0;
  for (int kt = 0; kt < ktiles; ++kt) {
    const bool haveNext = (kt + 1 < ktiles);
    if (haveNext) {
      gload(kt + 1);
      if (kt + 2 < ktiles) {  // pull tile kt+2 toward L2 (global_prefetch_b8)
        const int kk2 = (kt + 2) << 5;
        __builtin_prefetch(A + (size_t)(bm * 128 + aRow[0]) * lda + kk2 + aK8[0], 0, 1);
        __builtin_prefetch(Bm + (size_t)(kk2 + bKr[0]) * ldb + bn * 128 + bN8[0], 0, 1);
      }
    }
    const unsigned short* bufA = ldsA + cur * 5120;
    const unsigned short* bufB = ldsBT + cur * 5120;
    bf16x16 af[4], bfv[2];
#pragma unroll
    for (int ms = 0; ms < 4; ++ms)
      af[ms] = frag_a(bufA + (wm * 64 + ms * 16 + ln) * 40 + hf * 8);
#pragma unroll
    for (int ns = 0; ns < 2; ++ns)
      bfv[ns] = frag_b(bufB + (wn * 32 + ns * 16 + ln) * 40 + hf * 8);
#pragma unroll
    for (int ms = 0; ms < 4; ++ms)
#pragma unroll
      for (int ns = 0; ns < 2; ++ns)
        acc[ms][ns] = wmma_bf16(af[ms], bfv[ns], acc[ms][ns]);
    if (haveNext) {
      lstore(cur ^ 1);
      __syncthreads();
      cur ^= 1;
    }
  }

  // ---- epilogue ----
#pragma unroll
  for (int ms = 0; ms < 4; ++ms) {
#pragma unroll
    for (int ns = 0; ns < 2; ++ns) {
      const int col = bn * 128 + wn * 32 + ns * 16 + ln;
      const float bv = (col < nbias) ? bias[col] : 0.f;
#pragma unroll
      for (int r = 0; r < 8; ++r) {
        const int row = bm * 128 + wm * 64 + ms * 16 + hf * 8 + r;
        const float v = acc[ms][ns][r] + bv;
        const size_t o = (size_t)row * ldc + col;
        if (mode == 0) {
          ((float*)Cout)[o] = v;
        } else if (mode == 2) {
          ((unsigned short*)Cout)[o] = f2bf(v / (1.f + __expf(-v)));
        } else {
          ((float*)Cout)[o] = v + res[o];
        }
      }
    }
  }
}

// ---------------- LN + adaLN modulate: xn = LN(x)*(1+scale)+shift -> bf16 ---
__global__ __launch_bounds__(256) void modulate_kernel(
    const float* __restrict__ x, const float* __restrict__ se,
    const float* __restrict__ ln_w, const float* __restrict__ ln_b,
    unsigned short* __restrict__ xn) {
  const int row = blockIdx.x;
  const int tid = threadIdx.x;
  __shared__ float r1[256], r2[256];
  float s = 0.f, ss = 0.f;
  for (int c = tid; c < Cc; c += 256) {
    float v = x[(size_t)row * Cc + c];
    s += v; ss += v * v;
  }
  r1[tid] = s; r2[tid] = ss;
  __syncthreads();
  for (int st = 128; st > 0; st >>= 1) {
    if (tid < st) { r1[tid] += r1[tid + st]; r2[tid] += r2[tid + st]; }
    __syncthreads();
  }
  float mu = r1[0] / (float)Cc;
  float var = fmaxf(r2[0] / (float)Cc - mu * mu, 0.f);
  float rs = rsqrtf(var + EPSf);
  for (int c = tid; c < Cc; c += 256) {
    float v = x[(size_t)row * Cc + c];
    float n = (v - mu) * rs * ln_w[c] + ln_b[c];
    float sc = se[(size_t)row * (2 * Cc) + c];
    float sh = se[(size_t)row * (2 * Cc) + Cc + c];
    xn[(size_t)row * Cc + c] = f2bf(n * (1.f + sc) + sh);
  }
}

// ---------------- per-head q/k LayerNorm + pack (one wave per (b,n,h)) ------
__global__ __launch_bounds__(256) void qkln_kernel(
    const float* __restrict__ qkv, const float* __restrict__ qn_w,
    const float* __restrict__ qn_b, const float* __restrict__ kn_w,
    const float* __restrict__ kn_b, unsigned short* __restrict__ Qo,
    unsigned short* __restrict__ Ko, unsigned short* __restrict__ Vt,
    unsigned short* __restrict__ Qp) {
  const int w = blockIdx.x * 8 + (threadIdx.x >> 5);
  const int lane = threadIdx.x & 31;
  const int h = w & 15, n = (w >> 4) & 2047, b = w >> 15;
  const size_t rowbase = ((size_t)(b * Nn + n)) * (3 * Cc) + h * Dd;
  const int d0 = lane, d1 = lane + 32;

  float q0 = qkv[rowbase + d0], q1 = qkv[rowbase + d1];
  float k0 = qkv[rowbase + Cc + d0], k1 = qkv[rowbase + Cc + d1];
  float v0 = qkv[rowbase + 2 * Cc + d0], v1 = qkv[rowbase + 2 * Cc + d1];

  float s = q0 + q1;
  for (int off = 1; off < 32; off <<= 1) s += __shfl_xor(s, off, 32);
  float mu = s / 64.f;
  float a0 = q0 - mu, a1 = q1 - mu;
  float vs = a0 * a0 + a1 * a1;
  for (int off = 1; off < 32; off <<= 1) vs += __shfl_xor(vs, off, 32);
  float rs = rsqrtf(vs / 64.f + EPSf);
  float qo0 = a0 * rs * qn_w[d0] + qn_b[d0];
  float qo1 = a1 * rs * qn_w[d1] + qn_b[d1];

  s = k0 + k1;
  for (int off = 1; off < 32; off <<= 1) s += __shfl_xor(s, off, 32);
  mu = s / 64.f;
  a0 = k0 - mu; a1 = k1 - mu;
  vs = a0 * a0 + a1 * a1;
  for (int off = 1; off < 32; off <<= 1) vs += __shfl_xor(vs, off, 32);
  rs = rsqrtf(vs / 64.f + EPSf);
  float ko0 = a0 * rs * kn_w[d0] + kn_b[d0];
  float ko1 = a1 * rs * kn_w[d1] + kn_b[d1];

  const size_t bh = (size_t)(b * Hh + h);
  const size_t hrow = (bh * Nn + n) * Dd;
  Qo[hrow + d0] = f2bf(qo0); Qo[hrow + d1] = f2bf(qo1);
  Ko[hrow + d0] = f2bf(ko0); Ko[hrow + d1] = f2bf(ko1);
  const size_t prow = ((size_t)(b * Nn + n)) * Cc + h * Dd;
  Qp[prow + d0] = f2bf(qo0); Qp[prow + d1] = f2bf(qo1);
  Vt[(bh * Dd + d0) * Nn + n] = f2bf(v0);
  Vt[(bh * Dd + d1) * Nn + n] = f2bf(v1);
}

// ---------------- flash attention: O = softmax(QK^T/sqrt(D)) V --------------
__global__ __launch_bounds__(256) void attn_kernel(
    const unsigned short* __restrict__ Q, const unsigned short* __restrict__ Kb,
    const unsigned short* __restrict__ Vt, float* __restrict__ O) {
  __shared__ unsigned short ldsK[64 * 72];   // keys x D
  __shared__ unsigned short ldsV[64 * 72];   // D x keys (V transposed)
  __shared__ unsigned short ldsP[8][16 * 64];
  const int tid = threadIdx.x;
  const int wv = tid >> 5, lane = tid & 31;
  const int hf = lane >> 4, ln = lane & 15;
  const int bh = blockIdx.y;
  const int q0 = blockIdx.x * 128;
  const float SCALE = 0.125f;  // 1/sqrt(64)

  const size_t qbase = (((size_t)bh * Nn) + q0 + wv * 16 + ln) * Dd;
  bf16x16 qf[2];
#pragma unroll
  for (int kc = 0; kc < 2; ++kc) qf[kc] = frag_a(Q + qbase + kc * 32 + hf * 8);

  f32x8 oacc[4];
#pragma unroll
  for (int j = 0; j < 4; ++j)
#pragma unroll
    for (int e = 0; e < 8; ++e) oacc[j][e] = 0.f;
  float m[8], l[8];
#pragma unroll
  for (int r = 0; r < 8; ++r) { m[r] = -3.0e38f; l[r] = 0.f; }

  for (int kv = 0; kv < Nn; kv += 64) {
    __syncthreads();
#pragma unroll
    for (int c = 0; c < 2; ++c) {
      int chunk = tid * 2 + c;          // 0..511
      int row = chunk >> 3, c8 = (chunk & 7) * 8;
      *(uint4*)(ldsK + row * 72 + c8) =
          *(const uint4*)(Kb + (((size_t)bh * Nn) + kv + row) * Dd + c8);
      *(uint4*)(ldsV + row * 72 + c8) =
          *(const uint4*)(Vt + (((size_t)bh * Dd) + row) * Nn + kv + c8);
    }
    if (kv + 64 < Nn) {  // prefetch next K/V tile toward L2
      int row = (tid * 2) >> 3, c8 = ((tid * 2) & 7) * 8;
      __builtin_prefetch(Kb + (((size_t)bh * Nn) + kv + 64 + row) * Dd + c8, 0, 1);
      __builtin_prefetch(Vt + (((size_t)bh * Dd) + row) * Nn + kv + 64 + c8, 0, 1);
    }
    __syncthreads();

    f32x8 sacc[4];
#pragma unroll
    for (int j = 0; j < 4; ++j)
#pragma unroll
      for (int e = 0; e < 8; ++e) sacc[j][e] = 0.f;
#pragma unroll
    for (int ns = 0; ns < 4; ++ns)
#pragma unroll
      for (int kc = 0; kc < 2; ++kc) {
        bf16x16 kf = frag_b(ldsK + (ns * 16 + ln) * 72 + kc * 32 + hf * 8);
        sacc[ns] = wmma_bf16(qf[kc], kf, sacc[ns]);
      }

    // online softmax (rows = hf*8 + r, reduced across the 16-lane half)
    float mn[8];
#pragma unroll
    for (int r = 0; r < 8; ++r) mn[r] = m[r];
#pragma unroll
    for (int ns = 0; ns < 4; ++ns)
#pragma unroll
      for (int r = 0; r < 8; ++r) mn[r] = fmaxf(mn[r], sacc[ns][r] * SCALE);
    for (int off = 1; off < 16; off <<= 1)
#pragma unroll
      for (int r = 0; r < 8; ++r)
        mn[r] = fmaxf(mn[r], __shfl_xor(mn[r], off, 32));
    float corr[8], rsum[8];
#pragma unroll
    for (int r = 0; r < 8; ++r) {
      corr[r] = __expf(m[r] - mn[r]);
      m[r] = mn[r];
      rsum[r] = 0.f;
    }
#pragma unroll
    for (int ns = 0; ns < 4; ++ns)
#pragma unroll
      for (int r = 0; r < 8; ++r) {
        float p = __expf(sacc[ns][r] * SCALE - mn[r]);
        rsum[r] += p;
        ldsP[wv][(hf * 8 + r) * 64 + ns * 16 + ln] = f2bf(p);
      }
    for (int off = 1; off < 16; off <<= 1)
#pragma unroll
      for (int r = 0; r < 8; ++r) rsum[r] += __shfl_xor(rsum[r], off, 32);
#pragma unroll
    for (int r = 0; r < 8; ++r) l[r] = l[r] * corr[r] + rsum[r];
#pragma unroll
    for (int j = 0; j < 4; ++j)
#pragma unroll
      for (int r = 0; r < 8; ++r) oacc[j][r] *= corr[r];
    __syncthreads();

    bf16x16 pf[2];
#pragma unroll
    for (int kc = 0; kc < 2; ++kc)
      pf[kc] = frag_a(&ldsP[wv][ln * 64 + kc * 32 + hf * 8]);
#pragma unroll
    for (int d4 = 0; d4 < 4; ++d4)
#pragma unroll
      for (int kc = 0; kc < 2; ++kc) {
        bf16x16 vf = frag_b(ldsV + (d4 * 16 + ln) * 72 + kc * 32 + hf * 8);
        oacc[d4] = wmma_bf16(pf[kc], vf, oacc[d4]);
      }
  }

  float inv[8];
#pragma unroll
  for (int r = 0; r < 8; ++r) inv[r] = (l[r] > 0.f) ? 1.f / l[r] : 0.f;
#pragma unroll
  for (int d4 = 0; d4 < 4; ++d4)
#pragma unroll
    for (int r = 0; r < 8; ++r) {
      size_t row = (size_t)bh * Nn + q0 + wv * 16 + hf * 8 + r;
      O[row * Dd + d4 * 16 + ln] = oacc[d4][r] * inv[r];
    }
}

// ---------------- SwiGLU elementwise: hsw = silu(g) * v -> bf16 -------------
__global__ void hsw_kernel(const float* __restrict__ vg,
                           unsigned short* __restrict__ hsw) {
  const size_t total = (size_t)Mrows * INNER_PAD;
  size_t i = (size_t)blockIdx.x * blockDim.x + threadIdx.x;
  const size_t stride = (size_t)gridDim.x * blockDim.x;
  for (; i < total; i += stride) {
    size_t row = i / INNER_PAD;
    int j = (int)(i - row * INNER_PAD);
    if (j < INNERn) {
      float g = vg[row * (size_t)VG_N_PAD + INNERn + j];
      float vv = vg[row * (size_t)VG_N_PAD + j];
      hsw[i] = f2bf(vv * (g / (1.f + __expf(-g))));
    } else {
      hsw[i] = 0;
    }
  }
}

// ---------------- per-(b,h) RMS over (N,D) ----------------------------------
__global__ __launch_bounds__(256) void rms_kernel(const float* __restrict__ f,
                                                  float* __restrict__ rms) {
  const int bh = blockIdx.x, b = bh >> 4, h = bh & 15;
  float s = 0.f;
  for (int n = threadIdx.x; n < Nn; n += 256) {
    const float* p = f + ((size_t)(b * Nn + n)) * Cc + h * Dd;
    for (int d = 0; d < Dd; ++d) { float v = p[d]; s += v * v; }
  }
  __shared__ float red[256];
  red[threadIdx.x] = s;
  __syncthreads();
  for (int st = 128; st > 0; st >>= 1) {
    if (threadIdx.x < st) red[threadIdx.x] += red[threadIdx.x + st];
    __syncthreads();
  }
  if (threadIdx.x == 0)
    rms[bh] = fmaxf(sqrtf(red[0] / (float)(Nn * Dd)), 1e-6f);
}

// ---------------- gate = tanh(x @ W_g + b_g), stored (B,H,N) ----------------
__global__ __launch_bounds__(256) void gate_kernel(
    const float* __restrict__ x, const float* __restrict__ Wg,
    const float* __restrict__ bg, float* __restrict__ gate) {
  const int row = blockIdx.x, b = row >> 11, n = row & 2047;
  float part[16];
#pragma unroll
  for (int h = 0; h < 16; ++h) part[h] = 0.f;
  for (int k = threadIdx.x; k < Cc; k += 256) {
    float xv = x[(size_t)row * Cc + k];
#pragma unroll
    for (int h = 0; h < 16; ++h) part[h] += xv * Wg[k * 16 + h];
  }
  __shared__ float red[256];
  for (int h = 0; h < 16; ++h) {
    __syncthreads();
    red[threadIdx.x] = part[h];
    __syncthreads();
    for (int st = 128; st > 0; st >>= 1) {
      if (threadIdx.x < st) red[threadIdx.x] += red[threadIdx.x + st];
      __syncthreads();
    }
    if (threadIdx.x == 0)
      gate[((size_t)(b * Hh + h)) * Nn + n] = tanhf(red[0] + bg[h]);
  }
}

// ---------------- combine: oc = o + gate * f/rms -> bf16 (B,N,C) ------------
__global__ void combine_kernel(const float* __restrict__ o,
                               const float* __restrict__ f,
                               const float* __restrict__ rms,
                               const float* __restrict__ gate,
                               unsigned short* __restrict__ oc) {
  const size_t total = (size_t)Mrows * Cc;
  size_t i = (size_t)blockIdx.x * blockDim.x + threadIdx.x;
  const size_t stride = (size_t)gridDim.x * blockDim.x;
  for (; i < total; i += stride) {
    int c = (int)(i & 1023);
    size_t row = i >> 10;
    int n = (int)(row & 2047), b = (int)(row >> 11);
    int h = c >> 6, d = c & 63;
    int bh = b * Hh + h;
    float ov = o[(((size_t)bh) * Nn + n) * Dd + d];
    float fv = f[i] / rms[bh];
    float gv = gate[((size_t)bh) * Nn + n];
    oc[i] = f2bf(ov + gv * fv);
  }
}

// ---------------- host ----------------
extern "C" void kernel_launch(void* const* d_in, const int* in_sizes, int n_in,
                              void* d_out, int out_size, void* d_ws,
                              size_t ws_size, hipStream_t stream) {
  (void)in_sizes; (void)n_in; (void)out_size; (void)ws_size;
  const float* x     = (const float*)d_in[0];
  const float* emb   = (const float*)d_in[1];
  const float* W_emb = (const float*)d_in[2];
  const float* b_emb = (const float*)d_in[3];
  const float* ln_w  = (const float*)d_in[4];
  const float* ln_b  = (const float*)d_in[5];
  const float* W_f   = (const float*)d_in[6];
  const float* b_f   = (const float*)d_in[7];
  const float* qn_w  = (const float*)d_in[8];
  const float* qn_b  = (const float*)d_in[9];
  const float* kn_w  = (const float*)d_in[10];
  const float* kn_b  = (const float*)d_in[11];
  const float* W_ao  = (const float*)d_in[12];
  const float* b_ao  = (const float*)d_in[13];
  const float* W_mo  = (const float*)d_in[14];
  const float* b_mo  = (const float*)d_in[15];
  const float* Wm_in = (const float*)d_in[16];
  const float* bm_in = (const float*)d_in[17];
  const float* Wm_out= (const float*)d_in[18];
  const float* bm_out= (const float*)d_in[19];
  const float* W_g   = (const float*)d_in[20];
  const float* b_g   = (const float*)d_in[21];
  float* out = (float*)d_out;

  char* w = (char*)d_ws;
  size_t off = 0;
  auto take = [&](size_t bytes) {
    size_t o = off;
    off += (bytes + 255) & ~(size_t)255;
    return o;
  };
  // bf16 weight copies (padded where needed)
  const size_t oWEmb = take((size_t)1024 * 2048 * 2);
  const size_t oWF   = take((size_t)1024 * 7168 * 2);
  const size_t oWMin = take((size_t)1024 * VG_N_PAD * 2);     // N padded 5460->5504
  const size_t oWMout= take((size_t)INNER_PAD * 1024 * 2);    // K padded 2730->2752
  const size_t oWAo  = take((size_t)1024 * 1024 * 2);
  const size_t oWMo  = take((size_t)4096 * 1024 * 2);
  // overlay region: emb_bf / se / xn / qkv are dead before vg GEMM runs
  const size_t ovl   = off;
  const size_t oEmbB = take((size_t)8192 * 1024 * 2);
  const size_t oSe   = take((size_t)8192 * 2048 * 4);
  const size_t oXn   = take((size_t)8192 * 1024 * 2);
  const size_t oQkv  = take((size_t)8192 * 3072 * 4);
  const size_t oVg   = ovl;  // 8192*5504*4 fits inside the dead overlay
  // persistent
  const size_t oMsw  = take((size_t)8192 * 4096 * 2);
  const size_t oQb   = take((size_t)8192 * 1024 * 2);
  const size_t oKb   = take((size_t)8192 * 1024 * 2);
  const size_t oVtb  = take((size_t)8192 * 1024 * 2);
  const size_t oQp   = take((size_t)8192 * 1024 * 2);
  const size_t oO    = take((size_t)8192 * 1024 * 4);
  const size_t oHsw  = take((size_t)8192 * (size_t)INNER_PAD * 2);
  const size_t oF    = take((size_t)8192 * 1024 * 4);
  const size_t oRms  = take(256);
  const size_t oGate = take((size_t)8192 * 16 * 4);
  const size_t oOc   = take((size_t)8192 * 1024 * 2);
  const size_t oX2   = take((size_t)8192 * 1024 * 4);

  auto US = [&](size_t o) { return (unsigned short*)(w + o); };
  auto FP = [&](size_t o) { return (float*)(w + o); };

  // 1) f32 -> bf16 converts (weights + emb activations), padding where needed
  convert_pad_kernel<<<4096, 256, 0, stream>>>(W_emb, US(oWEmb), 1024, 2048, 2048, 1024);
  convert_pad_kernel<<<4096, 256, 0, stream>>>(W_f, US(oWF), 1024, 7168, 7168, 1024);
  convert_pad_kernel<<<4096, 256, 0, stream>>>(Wm_in, US(oWMin), 1024, 5460, VG_N_PAD, 1024);
  convert_pad_kernel<<<4096, 256, 0, stream>>>(Wm_out, US(oWMout), 2730, 1024, 1024, INNER_PAD);
  convert_pad_kernel<<<4096, 256, 0, stream>>>(W_ao, US(oWAo), 1024, 1024, 1024, 1024);
  convert_pad_kernel<<<4096, 256, 0, stream>>>(W_mo, US(oWMo), 4096, 1024, 1024, 4096);
  convert_pad_kernel<<<4096, 256, 0, stream>>>(emb, US(oEmbB), 8192, 1024, 1024, 8192);

  auto gemm = [&](const unsigned short* A, int lda, const unsigned short* Bm,
                  int ldb, const float* bias, int nbias, void* Cp, int ldc,
                  const float* resp, int M, int N, int K, int mode) {
    dim3 g(N / 128, M / 128);
    gemm_bf16_kernel<<<g, 256, 0, stream>>>(A, lda, Bm, ldb, bias, nbias, Cp,
                                            ldc, resp, M, N, K, mode);
  };

  // 2) se = emb @ W_emb + b_emb
  gemm(US(oEmbB), 1024, US(oWEmb), 2048, b_emb, 2048, FP(oSe), 2048, nullptr,
       8192, 2048, 1024, 0);
  // 3) xn = LN(x)*(1+scale)+shift (bf16)
  modulate_kernel<<<8192, 256, 0, stream>>>(x, FP(oSe), ln_w, ln_b, US(oXn));
  // 4) qkv = xn @ W_f[:, :3072] + b_f[:3072] (f32)
  gemm(US(oXn), 1024, US(oWF), 7168, b_f, 3072, FP(oQkv), 3072, nullptr,
       8192, 3072, 1024, 0);
  // 5) msw = silu(xn @ W_f[:, 3072:] + b_f[3072:]) (bf16)
  gemm(US(oXn), 1024, US(oWF) + 3072, 7168, b_f + 3072, 4096, US(oMsw), 4096,
       nullptr, 8192, 4096, 1024, 2);
  // 6) per-head q/k LN + packing (q,k: (B,H,N,D); v transposed (B,H,D,N); qp)
  qkln_kernel<<<16384, 256, 0, stream>>>(FP(oQkv), qn_w, qn_b, kn_w, kn_b,
                                         US(oQb), US(oKb), US(oVtb), US(oQp));
  // 7) flash attention
  {
    dim3 g(Nn / 128, Bb * Hh);
    attn_kernel<<<g, 256, 0, stream>>>(US(oQb), US(oKb), US(oVtb), FP(oO));
  }
  // 8) vg = qp @ Wm_in + bm_in (f32, N padded to 5504, overlay region)
  gemm(US(oQp), 1024, US(oWMin), VG_N_PAD, bm_in, 5460, FP(oVg), VG_N_PAD,
       nullptr, 8192, VG_N_PAD, 1024, 0);
  // 9) hsw = silu(g) * v (bf16, row stride padded to 2752)
  hsw_kernel<<<8192, 256, 0, stream>>>(FP(oVg), US(oHsw));
  // 10) f = hsw @ Wm_out + bm_out (f32, K padded to 2752 with zero rows)
  gemm(US(oHsw), INNER_PAD, US(oWMout), 1024, bm_out, 1024, FP(oF), 1024,
       nullptr, 8192, 1024, INNER_PAD, 0);
  // 11) per-(b,h) rms
  rms_kernel<<<64, 256, 0, stream>>>(FP(oF), FP(oRms));
  // 12) gate = tanh(x @ W_g + b_g)
  gate_kernel<<<8192, 256, 0, stream>>>(x, W_g, b_g, FP(oGate));
  // 13) oc = o + gate * f/rms (bf16, (B,N,C))
  combine_kernel<<<8192, 256, 0, stream>>>(FP(oO), FP(oF), FP(oRms), FP(oGate),
                                           US(oOc));
  // 14) x2 = x + oc @ W_ao + b_ao
  gemm(US(oOc), 1024, US(oWAo), 1024, b_ao, 1024, FP(oX2), 1024, x,
       8192, 1024, 1024, 3);
  // 15) out = x2 + msw @ W_mo + b_mo
  gemm(US(oMsw), 4096, US(oWMo), 1024, b_mo, 1024, out, 1024, FP(oX2),
       8192, 1024, 4096, 3);
}
